// QuantizedMatryoshkaDecoder_79130477462200
// MI455X (gfx1250) — 8-bit matrix operand order repaired and checked by reading, compiled, not run on hardware
//
#include <hip/hip_runtime.h>


typedef __attribute__((ext_vector_type(8))) int v8i;

#define BATCH      4096
#define IN_FEAT    8192
#define OUT_FEAT   512
#define NBITS      8
#define SLICE      1024
#define TOPK       32
#define RES_PLANE  (BATCH * OUT_FEAT)

__global__ void zero_out8(float* out) {
    if (threadIdx.x < NBITS) out[threadIdx.x] = 0.0f;
}

__global__ void pack_weights(const float* __restrict__ W,
                             const float* __restrict__ Wm,
                             signed char* __restrict__ Bpk) {
    int idx   = blockIdx.x * 256 + threadIdx.x;
    int t     = idx & 31;
    int lane  = (idx >> 5) & 31;
    int kb    = (idx >> 10) & 15;
    int ntile = (idx >> 14) & 31;
    int nb    = (idx >> 19) & 7;
    int v = t >> 2, j = t & 3;
    int klo = ((lane >> 4) << 4) | ((v >> 2) << 5) | ((v & 3) << 2) | j;
    int K   = (kb << 6) + klo;
    int col = (ntile << 4) + (lane & 15);
    int row = (nb << 10) + K;
    float w  = W [row * OUT_FEAT + col];
    float wm = Wm[row * OUT_FEAT + col];
    int s = ((w >= 0.0f) ? 1 : -1) + ((wm >= 0.0f) ? 1 : -1);
    Bpk[idx] = (signed char)(s >> 1);
}

__global__ void topk_sign(const float* __restrict__ latent,
                          signed char* __restrict__ A,
                          float* __restrict__ outLG) {
    const int lane  = threadIdx.x & 31;
    const int wave  = threadIdx.x >> 5;
    const int slice = blockIdx.x * 8 + wave;
    const int b  = slice >> 3;
    const int nb = slice & 7;
    const float* src = latent + (size_t)b * IN_FEAT + nb * SLICE + lane * 32;

    float    v[32];
    unsigned k[32];
#pragma unroll
    for (int q = 0; q < 8; ++q) {
        float4 f = ((const float4*)src)[q];
        v[q * 4 + 0] = f.x; v[q * 4 + 1] = f.y;
        v[q * 4 + 2] = f.z; v[q * 4 + 3] = f.w;
    }
#pragma unroll
    for (int j = 0; j < 32; ++j) {
        int bi = __float_as_int(v[j]);
        k[j] = (bi >= 0) ? ((unsigned)bi | 0x80000000u) : ~(unsigned)bi;
    }

    unsigned T = 0;
    for (int bit = 31; bit >= 0; --bit) {
        unsigned cand = T | (1u << bit);
        int c = 0;
#pragma unroll
        for (int j = 0; j < 32; ++j) c += (k[j] >= cand) ? 1 : 0;
        for (int off = 16; off > 0; off >>= 1) c += __shfl_down(c, off, 32);
        c = __shfl(c, 0, 32);
        if (c >= TOPK) T = cand;
    }

    int words[8];
    int nnz = 0;
#pragma unroll
    for (int wd = 0; wd < 8; ++wd) {
        int acc = 0;
#pragma unroll
        for (int t = 0; t < 4; ++t) {
            int j = wd * 4 + t;
            int sg = 0;
            if (k[j] >= T) sg = (v[j] > 0.0f) ? 1 : ((v[j] < 0.0f) ? -1 : 0);
            nnz += (sg != 0) ? 1 : 0;
            acc |= (sg & 0xff) << (8 * t);
        }
        words[wd] = acc;
    }
    signed char* dst = A + (size_t)b * IN_FEAT + nb * SLICE + lane * 32;
    ((int4*)dst)[0] = make_int4(words[0], words[1], words[2], words[3]);
    ((int4*)dst)[1] = make_int4(words[4], words[5], words[6], words[7]);

    for (int off = 16; off > 0; off >>= 1) nnz += __shfl_down(nnz, off, 32);
    if (lane == 0) atomicAdd(outLG + nb, (float)nnz * (1.0f / (float)BATCH));
}

__device__ inline v8i load_a_frag(const signed char* p) {
    int2 x0 = *(const int2*)(p);
    int2 x1 = *(const int2*)(p + 8);
    int2 x2 = *(const int2*)(p + 32);
    int2 x3 = *(const int2*)(p + 40);
    v8i r;
    r[0] = x0.x; r[1] = x0.y; r[2] = x1.x; r[3] = x1.y;
    r[4] = x2.x; r[5] = x2.y; r[6] = x3.x; r[7] = x3.y;
    return r;
}

__device__ inline v8i load_b_frag(const signed char* p) {
    int4 x0 = *(const int4*)(p);
    int4 x1 = *(const int4*)(p + 16);
    v8i r;
    r[0] = x0.x; r[1] = x0.y; r[2] = x0.z; r[3] = x0.w;
    r[4] = x1.x; r[5] = x1.y; r[6] = x1.z; r[7] = x1.w;
    return r;
}

__global__ __launch_bounds__(256) void
gemm_cumsum_wmma(const signed char* __restrict__ A,
                 const signed char* __restrict__ Bpk,
                 float* __restrict__ out) {
    const int lane = threadIdx.x & 31;
    const int w    = threadIdx.x >> 5;
    const int wm   = w & 3;
    const int wn   = w >> 2;
    const int m0   = blockIdx.y * 128 + wm * 32;
    const int n0   = blockIdx.x * 64  + wn * 32;

    const int mrow = lane & 15;
    const int khi  = (lane >> 4) << 4;

    float pre00[8], pre01[8], pre10[8], pre11[8];
#pragma unroll
    for (int e = 0; e < 8; ++e) { pre00[e] = pre01[e] = pre10[e] = pre11[e] = 0.0f; }

    float scale = 2.0f;
    for (int nb = 0; nb < NBITS; ++nb) {
        v8i acc00 = {0,0,0,0,0,0,0,0}, acc01 = {0,0,0,0,0,0,0,0};
        v8i acc10 = {0,0,0,0,0,0,0,0}, acc11 = {0,0,0,0,0,0,0,0};

        const signed char* A0 = A + (size_t)(m0 + mrow) * IN_FEAT + (nb << 10) + khi;
        const signed char* A1 = A0 + (size_t)16 * IN_FEAT;
        const signed char* B0 = Bpk + (size_t)(((nb << 5) + (n0 >> 4)) << 4) * 1024 + (lane << 5);
        const signed char* B1 = B0 + (16 << 10);

        for (int kb = 0; kb < 16; ++kb) {
            v8i a0 = load_a_frag(A0 + (kb << 6));
            v8i a1 = load_a_frag(A1 + (kb << 6));
            v8i b0 = load_b_frag(B0 + (kb << 10));
            v8i b1 = load_b_frag(B1 + (kb << 10));
            acc00 = __builtin_amdgcn_wmma_i32_16x16x64_iu8(true, a0, true, b0, acc00, false, false);
            acc01 = __builtin_amdgcn_wmma_i32_16x16x64_iu8(true, a0, true, b1, acc01, false, false);
            acc10 = __builtin_amdgcn_wmma_i32_16x16x64_iu8(true, a1, true, b0, acc10, false, false);
            acc11 = __builtin_amdgcn_wmma_i32_16x16x64_iu8(true, a1, true, b1, acc11, false, false);
        }

        float* res = out + NBITS + (size_t)nb * RES_PLANE;
        const int rbase = (lane >> 4) << 3;
        const int col   = lane & 15;
#pragma unroll
        for (int e = 0; e < 8; ++e) {
            pre00[e] += scale * (float)acc00[e];
            pre01[e] += scale * (float)acc01[e];
            pre10[e] += scale * (float)acc10[e];
            pre11[e] += scale * (float)acc11[e];
            res[(size_t)(m0      + rbase + e) * OUT_FEAT + (n0      + col)] = pre00[e];
            res[(size_t)(m0      + rbase + e) * OUT_FEAT + (n0 + 16 + col)] = pre01[e];
            res[(size_t)(m0 + 16 + rbase + e) * OUT_FEAT + (n0      + col)] = pre10[e];
            res[(size_t)(m0 + 16 + rbase + e) * OUT_FEAT + (n0 + 16 + col)] = pre11[e];
        }
        scale *= 0.5f;
    }
}

extern "C" void kernel_launch(void* const* d_in, const int* in_sizes, int n_in,
                              void* d_out, int out_size, void* d_ws, size_t ws_size,
                              hipStream_t stream) {
    const float* latent = (const float*)d_in[0];
    const float* W      = (const float*)d_in[1];
    const float* Wm     = (const float*)d_in[2];
    float* out = (float*)d_out;

    signed char* Bpk  = (signed char*)d_ws;
    signed char* Amat = Bpk + (4u << 20);

    zero_out8<<<1, 32, 0, stream>>>(out);
    pack_weights<<<16384, 256, 0, stream>>>(W, Wm, Bpk);
    topk_sign<<<4096, 256, 0, stream>>>(latent, Amat, out);
    dim3 grid(OUT_FEAT / 64, BATCH / 128);
    gemm_cumsum_wmma<<<grid, 256, 0, stream>>>(Amat, Bpk, out);
}
